// PointNet2_32598801777041
// MI455X (gfx1250) — matrix operand order repaired and checked by reading, not compiled, not run on hardware
//
#include <hip/hip_runtime.h>
#include <math.h>

// ---------------------------------------------------------------------------
// PointNet++ forward for MI455X (gfx1250, wave32).
// All pointwise-MLP / conv / head layers run through a WMMA f16->f32 GEMM
// (v_wmma_f32_16x16x32_f16) with 2x2 register blocking (32x32 macro-tile per
// wave) for 2x the FLOP/byte vs a single-tile wave. Irregular stages (FPS,
// ball query, grouping, 3-NN interpolation, maxpool) are scalar kernels.
// ---------------------------------------------------------------------------

typedef __attribute__((ext_vector_type(16))) _Float16 v16h;
typedef __attribute__((ext_vector_type(8)))  _Float16 v8h;
typedef __attribute__((ext_vector_type(8)))  float    v8f;

#define BATCH 4
#define N0PTS 16384

static __host__ __device__ inline int ceil32i(int x) { return (x + 31) & ~31; }

// ---------------------------------------------------------------------------
// Transpose (B,C,N) -> (B,N,C), also extract xyz channels 0..2
// ---------------------------------------------------------------------------
__global__ void k_transpose(const float* __restrict__ src, int Bc, int C, int N,
                            float* __restrict__ pts, float* __restrict__ l0) {
    int i = blockIdx.x * blockDim.x + threadIdx.x;
    if (i >= Bc * N) return;
    int b = i / N, n = i % N;
    const float* sb = src + (size_t)b * C * N;
    float* pr = pts + (size_t)i * C;
    for (int c = 0; c < C; ++c) pr[c] = sb[(size_t)c * N + n];
    float* lr = l0 + (size_t)i * 3;
    lr[0] = pr[0]; lr[1] = pr[1]; lr[2] = pr[2];
}

// ---------------------------------------------------------------------------
// Farthest point sampling: one block per batch, serial npoint loop with a
// block-wide argmax reduction (first-index tie break, like jnp.argmax).
// ---------------------------------------------------------------------------
__global__ void __launch_bounds__(512)
k_fps(const float* __restrict__ xyz, int N, int np,
      float* __restrict__ dist, int* __restrict__ out) {
    __shared__ float sv[512];
    __shared__ int   si[512];
    __shared__ int   sfar;
    int b = blockIdx.x, tid = threadIdx.x, bd = blockDim.x;
    const float* P = xyz + (size_t)b * N * 3;
    float* D = dist + (size_t)b * N;
    for (int i = tid; i < N; i += bd) D[i] = 1e10f;
    if (tid == 0) sfar = 0;
    __syncthreads();
    for (int it = 0; it < np; ++it) {
        int far = sfar;
        if (tid == 0) out[b * np + it] = far;
        float fx = P[far * 3 + 0], fy = P[far * 3 + 1], fz = P[far * 3 + 2];
        float best = -1.0f; int bi = 0;
        for (int i = tid; i < N; i += bd) {
            float dx = P[i * 3 + 0] - fx, dy = P[i * 3 + 1] - fy, dz = P[i * 3 + 2] - fz;
            float d  = dx * dx + dy * dy + dz * dz;
            float nd = fminf(D[i], d);
            D[i] = nd;
            if (nd > best) { best = nd; bi = i; }
        }
        sv[tid] = best; si[tid] = bi;
        __syncthreads();
        for (int off = bd >> 1; off > 0; off >>= 1) {
            if (tid < off) {
                float v2 = sv[tid + off]; int j2 = si[tid + off];
                if (v2 > sv[tid] || (v2 == sv[tid] && j2 < si[tid])) { sv[tid] = v2; si[tid] = j2; }
            }
            __syncthreads();
        }
        if (tid == 0) sfar = si[0];
        __syncthreads();
    }
}

// gather selected xyz
__global__ void k_gather3(const float* __restrict__ xyz, const int* __restrict__ idx,
                          int Bc, int Nin, int np, float* __restrict__ out) {
    int i = blockIdx.x * blockDim.x + threadIdx.x;
    if (i >= Bc * np) return;
    int b = i / np;
    int g = idx[i];
    const float* s = xyz + ((size_t)b * Nin + g) * 3;
    float* d = out + (size_t)i * 3;
    d[0] = s[0]; d[1] = s[1]; d[2] = s[2];
}

// ---------------------------------------------------------------------------
// Ball query: first nsample indices (ascending) with d^2 <= r^2, pad w/ first.
// ---------------------------------------------------------------------------
__global__ void k_ball(const float* __restrict__ xyz, const float* __restrict__ nxyz,
                       int Bc, int N, int np, int ns, float r2, int* __restrict__ out) {
    int i = blockIdx.x * blockDim.x + threadIdx.x;
    if (i >= Bc * np) return;
    int b = i / np;
    const float* q = nxyz + (size_t)i * 3;
    const float* base = xyz + (size_t)b * N * 3;
    int cnt = 0, first = 0;
    for (int j = 0; j < N; ++j) {
        float dx = base[j * 3 + 0] - q[0];
        float dy = base[j * 3 + 1] - q[1];
        float dz = base[j * 3 + 2] - q[2];
        float d2 = dx * dx + dy * dy + dz * dz;
        if (d2 <= r2) {
            if (cnt == 0) first = j;
            if (cnt < ns) out[(size_t)i * ns + cnt] = j;
            ++cnt;
            if (cnt >= ns) break;
        }
    }
    for (int k = cnt; k < ns; ++k) out[(size_t)i * ns + k] = first;
}

// ---------------------------------------------------------------------------
// Build grouped GEMM input rows: [rel_xyz(3) | feat(Cf)] -> f16, zero-pad Kpad
// ---------------------------------------------------------------------------
__global__ void k_group(const float* __restrict__ xyz, const float* __restrict__ nxyz,
                        const float* __restrict__ feat, const int* __restrict__ idx,
                        int Bc, int N, int np, int ns, int Cf, int Kpad,
                        _Float16* __restrict__ X) {
    int r = blockIdx.x * blockDim.x + threadIdx.x;
    int total = Bc * np * ns;
    if (r >= total) return;
    int s_ = r % ns;
    int q  = (r / ns) % np;
    int b  = r / (ns * np);
    int gi = idx[((size_t)(b * np + q)) * ns + s_];
    const float* P = xyz + ((size_t)b * N + gi) * 3;
    const float* Q = nxyz + ((size_t)(b * np + q)) * 3;
    _Float16* row = X + (size_t)r * Kpad;
    row[0] = (_Float16)(P[0] - Q[0]);
    row[1] = (_Float16)(P[1] - Q[1]);
    row[2] = (_Float16)(P[2] - Q[2]);
    const float* F = feat + ((size_t)b * N + gi) * Cf;
    for (int c = 0; c < Cf; ++c) row[3 + c] = (_Float16)F[c];
    for (int c = 3 + Cf; c < Kpad; ++c) row[c] = (_Float16)0.0f;
}

// ---------------------------------------------------------------------------
// 3-NN inverse-distance interpolation + concat -> f16 GEMM rows
// ---------------------------------------------------------------------------
__global__ void k_interp(const float* __restrict__ xyz1, const float* __restrict__ xyz2,
                         const float* __restrict__ p1, const float* __restrict__ p2,
                         int Bc, int N1, int N2, int C1, int C2, int Kpad,
                         _Float16* __restrict__ X) {
    int i = blockIdx.x * blockDim.x + threadIdx.x;
    if (i >= Bc * N1) return;
    int b = i / N1;
    const float* q = xyz1 + (size_t)i * 3;
    const float* base = xyz2 + (size_t)b * N2 * 3;
    float d0 = 1e30f, d1 = 1e30f, d2 = 1e30f;
    int i0 = 0, i1 = 0, i2 = 0;
    for (int j = 0; j < N2; ++j) {
        float dx = base[j * 3 + 0] - q[0];
        float dy = base[j * 3 + 1] - q[1];
        float dz = base[j * 3 + 2] - q[2];
        float d = dx * dx + dy * dy + dz * dz;
        if (d < d0)      { d2 = d1; i2 = i1; d1 = d0; i1 = i0; d0 = d; i0 = j; }
        else if (d < d1) { d2 = d1; i2 = i1; d1 = d; i1 = j; }
        else if (d < d2) { d2 = d; i2 = j; }
    }
    float r0 = 1.0f / (d0 + 1e-8f), r1 = 1.0f / (d1 + 1e-8f), r2 = 1.0f / (d2 + 1e-8f);
    float inv = 1.0f / (r0 + r1 + r2);
    float w0 = r0 * inv, w1 = r1 * inv, w2 = r2 * inv;
    _Float16* row = X + (size_t)i * Kpad;
    int c = 0;
    if (p1) {
        const float* F1 = p1 + (size_t)i * C1;
        for (int k = 0; k < C1; ++k) row[c++] = (_Float16)F1[k];
    }
    const float* F2 = p2 + (size_t)((size_t)b * N2) * C2;
    for (int k = 0; k < C2; ++k) {
        float v = w0 * F2[(size_t)i0 * C2 + k] + w1 * F2[(size_t)i1 * C2 + k]
                + w2 * F2[(size_t)i2 * C2 + k];
        row[c++] = (_Float16)v;
    }
    for (; c < Kpad; ++c) row[c] = (_Float16)0.0f;
}

// ---------------------------------------------------------------------------
// Weight packing: W (cout x cin, f32) -> per-K-step WMMA B-tiles, f16.
// B (32x16 K x N) lane layout: lane = n%16 + 16*g ; lane holds K = 8g+e (e<8), 16+8g+(e-8) (e>=8).
// Linear index: ((kt*Ntiles + nt)*32 + lane)*16 + e
// ---------------------------------------------------------------------------
__global__ void k_prep_w(const float* __restrict__ W, int cout, int cin,
                         int Kpad, int Npad, _Float16* __restrict__ Wp) {
    int tid = blockIdx.x * blockDim.x + threadIdx.x;
    int Ntiles = Npad >> 4;
    int total = (Kpad >> 5) * Ntiles * 32;
    if (tid >= total) return;
    int lane = tid & 31;
    int t2 = tid >> 5;
    int nt = t2 % Ntiles;
    int kt = t2 / Ntiles;
    int n = nt * 16 + (lane & 15);
    int g = lane >> 4;
    _Float16* dst = Wp + (size_t)tid * 16;
    for (int e = 0; e < 16; ++e) {
        int k = kt * 32 + (e < 8 ? g * 8 + e : 16 + g * 8 + (e - 8));
        float v = (n < cout && k < cin) ? W[(size_t)n * cin + k] : 0.0f;
        dst[e] = (_Float16)v;
    }
}

// fold BN into scale/shift: y = acc*s + t  (linear layers: s=1, t=bias)
__global__ void k_prep_scale(const float* __restrict__ b, const float* __restrict__ gamma,
                             const float* __restrict__ beta, const float* __restrict__ mean,
                             const float* __restrict__ var, int cout,
                             float* __restrict__ S, float* __restrict__ T) {
    int n = blockIdx.x * blockDim.x + threadIdx.x;
    if (n >= cout) return;
    if (gamma) {
        float sc = gamma[n] * rsqrtf(var[n] + 1e-5f);
        S[n] = sc;
        T[n] = (b[n] - mean[n]) * sc + beta[n];
    } else {
        S[n] = 1.0f;
        T[n] = b[n];
    }
}

__device__ inline float apply_act(float v, int act) {
    if (act == 1)      v = fmaxf(v, 0.0f);
    else if (act == 2) v = (v > 0.0f) ? v : 0.1f * v;
    else if (act == 3) v = 1.0f / (1.0f + __expf(-v));
    return v;
}

// ---------------------------------------------------------------------------
// WMMA GEMM, 2x2 register-blocked: each wave computes a 32x32 output macro-
// tile (4 accumulators), reusing each A/B fragment across 2 WMMAs -> 2x the
// FLOP per byte pulled from L2 vs a single-tile wave.
// A layout (16-bit, 16x32): lane<16 row=lane K{0..7,16..23}; lane>=16 K{8..15,24..31}
// C/D layout: lane l: n = l&15, VGPR r: m = r + 8*(l>>4)
// act: 0=linear 1=relu 2=leaky(0.1) 3=sigmoid
// ---------------------------------------------------------------------------
__global__ void __launch_bounds__(128)
k_gemm(const _Float16* __restrict__ X, const _Float16* __restrict__ Wp,
       const float* __restrict__ S, const float* __restrict__ T,
       _Float16* __restrict__ Yh, float* __restrict__ Yf,
       int rows, int Kpad, int Npad, int Ntrue, int act) {
    int lane = threadIdx.x & 31;
    int wave = blockIdx.x * (blockDim.x >> 5) + (threadIdx.x >> 5);
    int Nt2 = Npad >> 5;   // 32-wide column macro-tiles
    int Mt2 = rows >> 5;   // 32-tall row macro-tiles
    if (wave >= Mt2 * Nt2) return;   // wave-uniform: EXEC stays all-ones
    int mt = wave / Nt2, nt = wave % Nt2;
    int g = lane >> 4;
    int lrow0 = mt * 32 + (lane & 15);
    const _Float16* xrow0 = X + (size_t)lrow0 * Kpad + 8 * g;
    const _Float16* xrow1 = xrow0 + (size_t)16 * Kpad;
    const _Float16* wb0 = Wp + ((size_t)(2 * nt) * 32 + lane) * 16;
    const _Float16* wb1 = wb0 + 32 * 16;
    size_t wstride = (size_t)(Npad >> 4) * 32 * 16;   // per k-step
    v8f c00 = {0.f, 0.f, 0.f, 0.f, 0.f, 0.f, 0.f, 0.f};
    v8f c01 = c00, c10 = c00, c11 = c00;
    int Ksteps = Kpad >> 5;
    for (int kk = 0; kk < Ksteps; ++kk) {
        v8h lo0 = *(const v8h*)(xrow0 + (size_t)kk * 32);
        v8h hi0 = *(const v8h*)(xrow0 + (size_t)kk * 32 + 16);
        v8h lo1 = *(const v8h*)(xrow1 + (size_t)kk * 32);
        v8h hi1 = *(const v8h*)(xrow1 + (size_t)kk * 32 + 16);
        v16h a0 = __builtin_shufflevector(lo0, hi0, 0, 1, 2, 3, 4, 5, 6, 7,
                                                    8, 9, 10, 11, 12, 13, 14, 15);
        v16h a1 = __builtin_shufflevector(lo1, hi1, 0, 1, 2, 3, 4, 5, 6, 7,
                                                    8, 9, 10, 11, 12, 13, 14, 15);
        v16h b0 = *(const v16h*)(wb0 + (size_t)kk * wstride);
        v16h b1 = *(const v16h*)(wb1 + (size_t)kk * wstride);
        __builtin_prefetch(xrow0 + (size_t)(kk + 1) * 32, 0, 1);
        __builtin_prefetch(xrow1 + (size_t)(kk + 1) * 32, 0, 1);
        c00 = __builtin_amdgcn_wmma_f32_16x16x32_f16(false, a0, false, b0, (short)0, c00, false, false);
        c01 = __builtin_amdgcn_wmma_f32_16x16x32_f16(false, a0, false, b1, (short)0, c01, false, false);
        c10 = __builtin_amdgcn_wmma_f32_16x16x32_f16(false, a1, false, b0, (short)0, c10, false, false);
        c11 = __builtin_amdgcn_wmma_f32_16x16x32_f16(false, a1, false, b1, (short)0, c11, false, false);
    }
    int n0 = nt * 32 + (lane & 15);
    int n1 = n0 + 16;
    float sc0 = (n0 < Ntrue) ? S[n0] : 0.0f;
    float tb0 = (n0 < Ntrue) ? T[n0] : 0.0f;
    float sc1 = (n1 < Ntrue) ? S[n1] : 0.0f;
    float tb1 = (n1 < Ntrue) ? T[n1] : 0.0f;
    for (int r = 0; r < 8; ++r) {
        int m0 = mt * 32 + r + 8 * g;
        int m1 = m0 + 16;
        float v00 = apply_act(c00[r] * sc0 + tb0, act);
        float v01 = apply_act(c01[r] * sc1 + tb1, act);
        float v10 = apply_act(c10[r] * sc0 + tb0, act);
        float v11 = apply_act(c11[r] * sc1 + tb1, act);
        if (Yh) {
            Yh[(size_t)m0 * Npad + n0] = (n0 < Ntrue) ? (_Float16)v00 : (_Float16)0.0f;
            Yh[(size_t)m0 * Npad + n1] = (n1 < Ntrue) ? (_Float16)v01 : (_Float16)0.0f;
            Yh[(size_t)m1 * Npad + n0] = (n0 < Ntrue) ? (_Float16)v10 : (_Float16)0.0f;
            Yh[(size_t)m1 * Npad + n1] = (n1 < Ntrue) ? (_Float16)v11 : (_Float16)0.0f;
        }
        if (Yf) {
            if (n0 < Ntrue) {
                Yf[(size_t)m0 * Ntrue + n0] = v00;
                Yf[(size_t)m1 * Ntrue + n0] = v10;
            }
            if (n1 < Ntrue) {
                Yf[(size_t)m0 * Ntrue + n1] = v01;
                Yf[(size_t)m1 * Ntrue + n1] = v11;
            }
        }
    }
}

// max over nsample dimension, f16 -> f32 pooled features
__global__ void k_maxpool(const _Float16* __restrict__ Y, int groups, int ns,
                          int C, int Npad, float* __restrict__ out) {
    int i = blockIdx.x * blockDim.x + threadIdx.x;
    if (i >= groups * C) return;
    int c = i % C, gid = i / C;
    const _Float16* base = Y + (size_t)gid * ns * Npad + c;
    float m = -1e30f;
    for (int s = 0; s < ns; ++s) m = fmaxf(m, (float)base[(size_t)s * Npad]);
    out[(size_t)gid * C + c] = m;
}

// ---------------------------------------------------------------------------
// Host driver
// ---------------------------------------------------------------------------
struct LInfo { int base; int bn; int cin; int cout; };
// d_in layout assumption: d_in[0]=xyz, then params flattened jax-tree style
// (sorted dict keys): conv1(6), conv2(2), fp1(18), fp2(12), fp3(12), fp4(12),
// head_center(2), head_mlp(2), head_softmax(2), head_var(2), sa1..sa4(18 each).
// BN-layer leaf order (sorted): W, b, beta, gamma, mean, var.
static const LInfo g_layers[27] = {
    {71, 1, 12, 32},  {77, 1, 32, 32},  {83, 1, 32, 64},      // sa1: 0..2
    {89, 1, 67, 64},  {95, 1, 64, 64},  {101, 1, 64, 128},    // sa2: 3..5
    {107, 1, 131, 128},{113, 1, 128, 128},{119, 1, 128, 256}, // sa3: 6..8
    {125, 1, 259, 256},{131, 1, 256, 256},{137, 1, 256, 512}, // sa4: 9..11
    {51, 1, 768, 256}, {57, 1, 256, 256},                     // fp4: 12..13
    {39, 1, 384, 256}, {45, 1, 256, 256},                     // fp3: 14..15
    {27, 1, 320, 256}, {33, 1, 256, 128},                     // fp2: 16..17
    {9, 1, 128, 128},  {15, 1, 128, 128},{21, 1, 128, 128},   // fp1: 18..20
    {1, 1, 128, 128},                                         // conv1: 21
    {7, 0, 128, 256},                                         // conv2: 22
    {65, 0, 256, 256},                                        // head_mlp: 23
    {63, 0, 256, 1},                                          // head_center: 24
    {69, 0, 256, 260},                                        // head_var: 25
    {67, 0, 256, 19},                                         // head_softmax: 26
};

struct GW { _Float16* Wp; float* S; float* T; int cin, cout, Kpad, Npad; };

extern "C" void kernel_launch(void* const* d_in, const int* in_sizes, int n_in,
                              void* d_out, int out_size, void* d_ws, size_t ws_size,
                              hipStream_t stream) {
    (void)in_sizes; (void)out_size; (void)ws_size;
    if (!d_in || !d_out || !d_ws || n_in < 143) return;

    const int B = BATCH, N0 = N0PTS;
    size_t cur = 0;
    auto alloc = [&](size_t bytes) -> char* {
        char* p = (char*)d_ws + cur;
        cur += (bytes + 255) & ~(size_t)255;
        return p;
    };

    _Float16* PP0 = (_Float16*)alloc((size_t)B * N0 * 256 * 2);   // 32 MB ping
    _Float16* PP1 = (_Float16*)alloc((size_t)B * N0 * 256 * 2);   // 32 MB pong
    float* pts    = (float*)alloc((size_t)B * N0 * 9 * 4);
    float* l0xyz  = (float*)alloc((size_t)B * N0 * 3 * 4);
    float* l1xyz  = (float*)alloc((size_t)B * 1024 * 3 * 4);
    float* l2xyz  = (float*)alloc((size_t)B * 256 * 3 * 4);
    float* l3xyz  = (float*)alloc((size_t)B * 64 * 3 * 4);
    float* l4xyz  = (float*)alloc((size_t)B * 16 * 3 * 4);
    float* dist   = (float*)alloc((size_t)B * N0 * 4);
    int*   fpsidx = (int*)alloc((size_t)B * 1024 * 4);
    int*   ballix = (int*)alloc((size_t)B * 1024 * 32 * 4);
    float* l1f    = (float*)alloc((size_t)B * 1024 * 64 * 4);
    float* l2f    = (float*)alloc((size_t)B * 256 * 128 * 4);
    float* l3f    = (float*)alloc((size_t)B * 64 * 256 * 4);
    float* l4f    = (float*)alloc((size_t)B * 16 * 512 * 4);
    float* fp4f   = (float*)alloc((size_t)B * 64 * 256 * 4);
    float* fp3f   = (float*)alloc((size_t)B * 256 * 256 * 4);
    float* fp2f   = (float*)alloc((size_t)B * 1024 * 128 * 4);

    GW gw[27];
    for (int l = 0; l < 27; ++l) {
        const LInfo& li = g_layers[l];
        gw[l].cin = li.cin; gw[l].cout = li.cout;
        gw[l].Kpad = ceil32i(li.cin);
        gw[l].Npad = ceil32i(li.cout);
        gw[l].Wp = (_Float16*)alloc((size_t)gw[l].Kpad * gw[l].Npad * 2);
        gw[l].S  = (float*)alloc((size_t)gw[l].Npad * 4);
        gw[l].T  = (float*)alloc((size_t)gw[l].Npad * 4);
    }

    // ---- weight / scale prep (deterministic, runs every call) ----
    for (int l = 0; l < 27; ++l) {
        const LInfo& li = g_layers[l];
        const float* W = (const float*)d_in[li.base + 0];
        const float* b = (const float*)d_in[li.base + 1];
        const float* beta = li.bn ? (const float*)d_in[li.base + 2] : nullptr;
        const float* gamma = li.bn ? (const float*)d_in[li.base + 3] : nullptr;
        const float* mean = li.bn ? (const float*)d_in[li.base + 4] : nullptr;
        const float* var = li.bn ? (const float*)d_in[li.base + 5] : nullptr;
        int tot = (gw[l].Kpad >> 5) * (gw[l].Npad >> 4) * 32;
        k_prep_w<<<(tot + 255) / 256, 256, 0, stream>>>(W, li.cout, li.cin,
                                                        gw[l].Kpad, gw[l].Npad, gw[l].Wp);
        k_prep_scale<<<(li.cout + 255) / 256, 256, 0, stream>>>(b, gamma, beta, mean, var,
                                                                li.cout, gw[l].S, gw[l].T);
    }

    auto gemm = [&](const _Float16* X, const GW& w, _Float16* Yh, float* Yf,
                    int rows, int act) {
        int tiles = (rows >> 5) * (w.Npad >> 5);   // 32x32 macro-tiles
        int thr = tiles * 32;
        k_gemm<<<(thr + 127) / 128, 128, 0, stream>>>(X, w.Wp, w.S, w.T, Yh, Yf,
                                                      rows, w.Kpad, w.Npad, w.cout, act);
    };

    auto run_sa = [&](const float* xyz_in, int Nin, const float* feat, int Cf,
                      int np, float radius, int l0i, float* xyz_out, float* fout) {
        k_fps<<<B, 512, 0, stream>>>(xyz_in, Nin, np, dist, fpsidx);
        int tq = B * np;
        k_gather3<<<(tq + 255) / 256, 256, 0, stream>>>(xyz_in, fpsidx, B, Nin, np, xyz_out);
        k_ball<<<(tq + 255) / 256, 256, 0, stream>>>(xyz_in, xyz_out, B, Nin, np, 32,
                                                     radius * radius, ballix);
        int rows = tq * 32;
        k_group<<<(rows + 255) / 256, 256, 0, stream>>>(xyz_in, xyz_out, feat, ballix,
                                                        B, Nin, np, 32, Cf,
                                                        gw[l0i].Kpad, PP0);
        gemm(PP0, gw[l0i + 0], PP1, nullptr, rows, 1);
        gemm(PP1, gw[l0i + 1], PP0, nullptr, rows, 1);
        gemm(PP0, gw[l0i + 2], PP1, nullptr, rows, 1);
        int C = gw[l0i + 2].cout, Npad = gw[l0i + 2].Npad;
        int tt = tq * C;
        k_maxpool<<<(tt + 255) / 256, 256, 0, stream>>>(PP1, tq, 32, C, Npad, fout);
    };

    auto run_fp = [&](const float* xyz1, int N1, const float* xyz2, int N2,
                      const float* p1, int C1, const float* p2, int C2,
                      int li, int nl, float* fout) -> _Float16* {
        int ti = B * N1;
        k_interp<<<(ti + 255) / 256, 256, 0, stream>>>(xyz1, xyz2, p1, p2, B, N1, N2,
                                                       C1, C2, gw[li].Kpad, PP0);
        _Float16* cx = PP0; _Float16* cy = PP1;
        for (int l = 0; l < nl; ++l) {
            gemm(cx, gw[li + l], cy, (l == nl - 1) ? fout : nullptr, ti, 1);
            _Float16* t = cx; cx = cy; cy = t;
        }
        return cx;  // final activations (f16)
    };

    // ---- pipeline ----
    k_transpose<<<(B * N0 + 255) / 256, 256, 0, stream>>>((const float*)d_in[0], B, 9, N0,
                                                          pts, l0xyz);
    run_sa(l0xyz, N0,   pts, 9,    1024, 0.1f, 0,  l1xyz, l1f);
    run_sa(l1xyz, 1024, l1f, 64,   256,  0.2f, 3,  l2xyz, l2f);
    run_sa(l2xyz, 256,  l2f, 128,  64,   0.4f, 6,  l3xyz, l3f);
    run_sa(l3xyz, 64,   l3f, 256,  16,   0.8f, 9,  l4xyz, l4f);

    run_fp(l3xyz, 64,   l4xyz, 16,   l3f, 256, l4f,  512, 12, 2, fp4f);
    run_fp(l2xyz, 256,  l3xyz, 64,   l2f, 128, fp4f, 256, 14, 2, fp3f);
    run_fp(l1xyz, 1024, l2xyz, 256,  l1f, 64,  fp3f, 256, 16, 2, fp2f);
    _Float16* l0act = run_fp(l0xyz, N0, l1xyz, 1024, nullptr, 0, fp2f, 128, 18, 3, nullptr);
    _Float16* other = (l0act == PP0) ? PP1 : PP0;

    const int BN = B * N0;  // 65536
    float* out_s = (float*)d_out;
    float* out_c = out_s + (size_t)BN * 19;
    float* out_v = out_c + (size_t)BN * 1;
    float* out_f = out_v + (size_t)BN * 260;

    gemm(l0act, gw[21], other, nullptr, BN, 1);   // conv1: BN+relu
    gemm(other, gw[22], l0act, nullptr, BN, 0);   // conv2: linear
    gemm(l0act, gw[23], other, out_f, BN, 2);     // head_mlp: leaky -> f
    gemm(other, gw[24], nullptr, out_c, BN, 3);   // head_center: sigmoid
    gemm(other, gw[25], nullptr, out_v, BN, 1);   // head_var: relu(leaky)==relu
    gemm(other, gw[26], nullptr, out_s, BN, 2);   // head_softmax: leaky
}
